// AttnGRU_VNMT_17008070492417
// MI455X (gfx1250) — compile-verified, matrix operand order checked by reading, not run on hardware
//
#include <hip/hip_runtime.h>
#include <hip/hip_bf16.h>
#include <stdint.h>
#include <stddef.h>

// Problem constants (from reference setup_inputs)
#define B_ 32
#define S_ 64
#define V_ 32000
#define E_ 512
#define H_ 1024
#define G_ (3 * H_)

typedef __attribute__((ext_vector_type(16))) __bf16 v16bf;
typedef __attribute__((ext_vector_type(8)))  __bf16 v8bf;
typedef __attribute__((ext_vector_type(8)))  float  v8f;
typedef unsigned short u16b;

struct U128 { unsigned u[4]; };

// ---------------------------------------------------------------- utilities

__device__ __forceinline__ u16b f2bf_u(float f) {
    unsigned u = __builtin_bit_cast(unsigned, f);
    unsigned r = u + 0x7FFFu + ((u >> 16) & 1u);   // round-to-nearest-even
    return (u16b)(r >> 16);
}

// Load one 16x32 bf16 WMMA operand fragment (A row-major M x K, or W row-major
// N x K used as B). Per CDNA5 ISA layout: lane holds row/col = lane%16, K
// elements {0..7,16..23} for lanes 0-15 and {8..15,24..31} for lanes 16-31.
__device__ __forceinline__ v16bf load_frag(const u16b* __restrict__ rowPtr,
                                           int kBase, int lane) {
    const int koff = (lane & 16) ? 8 : 0;
    U128 lo = *(const U128*)(rowPtr + kBase + koff);
    U128 hi = *(const U128*)(rowPtr + kBase + 16 + koff);
    v8bf l = __builtin_bit_cast(v8bf, lo);
    v8bf h = __builtin_bit_cast(v8bf, hi);
    v16bf f;
#pragma unroll
    for (int i = 0; i < 8; ++i) { f[i] = l[i]; f[i + 8] = h[i]; }
    return f;
}

__device__ __forceinline__ v8f wmma_bf16(v16bf a, v16bf b, v8f c) {
    // 8 args: (neg_a, A, neg_b, B, c_mod, C, reuse_a, reuse_b)
    return __builtin_amdgcn_wmma_f32_16x16x32_bf16(false, a, false, b,
                                                   (short)0, c, false, false);
}

__device__ __forceinline__ void store_tile(float* __restrict__ C, v8f acc,
                                           int mBase, int rOff, int nCol, int N,
                                           const float* __restrict__ bias) {
    const float bv = bias ? bias[nCol] : 0.0f;
#pragma unroll
    for (int r = 0; r < 8; ++r)
        C[(size_t)(mBase + r + rOff) * N + nCol] = acc[r] + bv;
}

// ------------------------------------------------------------- GEMM (WMMA)
// C[M,N] = A[M,K](bf16) * W[N,K]^T(bf16) + bias[N]; f32 accumulate.
// Grid: (N/256, M/16), block = 128 threads (4 waves, each a 16x64 strip).
// Double-buffered register rotation, K advanced 64/iteration with the last
// 64-wide step peeled: every fragment load's destination IS the loop-carried
// value (no conditional assigns -> no v_mov rotation), so one fragment set's
// loads are always in flight behind the other set's WMMAs.
// Requires K % 64 == 0 (K = 512 or 1024 here).
__global__ __launch_bounds__(128)
void gemm_bf16_kernel(const u16b* __restrict__ A, const u16b* __restrict__ W,
                      const float* __restrict__ bias, float* __restrict__ C,
                      int M, int N, int K) {
    const int lane  = threadIdx.x & 31;
    const int wave  = threadIdx.x >> 5;
    const int col   = lane & 15;
    const int mBase = blockIdx.y << 4;
    const int nBase = (blockIdx.x << 8) + (wave << 6);

    const u16b* aRow = A + (size_t)(mBase + col) * K;
    const u16b* w0 = W + (size_t)(nBase +  0 + col) * K;
    const u16b* w1 = W + (size_t)(nBase + 16 + col) * K;
    const u16b* w2 = W + (size_t)(nBase + 32 + col) * K;
    const u16b* w3 = W + (size_t)(nBase + 48 + col) * K;

    v8f acc0 = {}, acc1 = {}, acc2 = {}, acc3 = {};

    // prologue: even set at k = 0
    v16bf ae  = load_frag(aRow, 0, lane);
    v16bf be0 = load_frag(w0, 0, lane);
    v16bf be1 = load_frag(w1, 0, lane);
    v16bf be2 = load_frag(w2, 0, lane);
    v16bf be3 = load_frag(w3, 0, lane);

    int k = 0;
    for (; k + 64 < K; k += 64) {
        // odd set (k+32) issues ahead of the even WMMAs
        v16bf ao  = load_frag(aRow, k + 32, lane);
        v16bf bo0 = load_frag(w0, k + 32, lane);
        v16bf bo1 = load_frag(w1, k + 32, lane);
        v16bf bo2 = load_frag(w2, k + 32, lane);
        v16bf bo3 = load_frag(w3, k + 32, lane);
        __builtin_prefetch(aRow + k + 128, 0, 3);
        __builtin_prefetch(w0 + k + 128, 0, 3);
        __builtin_prefetch(w1 + k + 128, 0, 3);
        __builtin_prefetch(w2 + k + 128, 0, 3);
        __builtin_prefetch(w3 + k + 128, 0, 3);
        acc0 = wmma_bf16(ae, be0, acc0);
        acc1 = wmma_bf16(ae, be1, acc1);
        acc2 = wmma_bf16(ae, be2, acc2);
        acc3 = wmma_bf16(ae, be3, acc3);
        // even reload (k+64) issues ahead of the odd WMMAs
        ae  = load_frag(aRow, k + 64, lane);
        be0 = load_frag(w0, k + 64, lane);
        be1 = load_frag(w1, k + 64, lane);
        be2 = load_frag(w2, k + 64, lane);
        be3 = load_frag(w3, k + 64, lane);
        acc0 = wmma_bf16(ao, bo0, acc0);
        acc1 = wmma_bf16(ao, bo1, acc1);
        acc2 = wmma_bf16(ao, bo2, acc2);
        acc3 = wmma_bf16(ao, bo3, acc3);
    }
    // peeled final 64-wide step (k = K - 64)
    {
        v16bf ao  = load_frag(aRow, k + 32, lane);
        v16bf bo0 = load_frag(w0, k + 32, lane);
        v16bf bo1 = load_frag(w1, k + 32, lane);
        v16bf bo2 = load_frag(w2, k + 32, lane);
        v16bf bo3 = load_frag(w3, k + 32, lane);
        acc0 = wmma_bf16(ae, be0, acc0);
        acc1 = wmma_bf16(ae, be1, acc1);
        acc2 = wmma_bf16(ae, be2, acc2);
        acc3 = wmma_bf16(ae, be3, acc3);
        acc0 = wmma_bf16(ao, bo0, acc0);
        acc1 = wmma_bf16(ao, bo1, acc1);
        acc2 = wmma_bf16(ao, bo2, acc2);
        acc3 = wmma_bf16(ao, bo3, acc3);
    }

    const int rOff = (lane & 16) ? 8 : 0;
    store_tile(C, acc0, mBase, rOff, nBase +  0 + col, N, bias);
    store_tile(C, acc1, mBase, rOff, nBase + 16 + col, N, bias);
    store_tile(C, acc2, mBase, rOff, nBase + 32 + col, N, bias);
    store_tile(C, acc3, mBase, rOff, nBase + 48 + col, N, bias);
}

// ------------------------------------------------------------ small kernels

__global__ void f2bf_kernel(const float* __restrict__ in, u16b* __restrict__ out,
                            size_t n) {
    size_t i = (size_t)blockIdx.x * blockDim.x + threadIdx.x;
    if (i < n) out[i] = f2bf_u(in[i]);
}

// attn_W (H x 2H) -> W_h = attn_W[:, :H], W_e = attn_W[:, H:], both bf16 (H x H)
__global__ void split_attn_kernel(const float* __restrict__ attnW,
                                  u16b* __restrict__ Wh, u16b* __restrict__ We) {
    size_t i = (size_t)blockIdx.x * blockDim.x + threadIdx.x;
    if (i >= (size_t)H_ * H_) return;
    int r = (int)(i / H_), c = (int)(i % H_);
    Wh[i] = f2bf_u(attnW[(size_t)r * (2 * H_) + c]);
    We[i] = f2bf_u(attnW[(size_t)r * (2 * H_) + H_ + c]);
}

// Gather embedding rows into (S,B,E) bf16; shift=1 reproduces emb_t_shift.
__global__ void gather_embed_kernel(const float* __restrict__ emb,
                                    const int* __restrict__ tok,
                                    u16b* __restrict__ out, int shift) {
    int sb = blockIdx.x;               // 0 .. S*B-1
    int s = sb / B_, b = sb % B_;
    bool zero = false;
    int idx = 0;
    if (shift) { if (s == 0) zero = true; else idx = tok[b * S_ + (s - 1)]; }
    else       { idx = tok[b * S_ + s]; }
    const float* src = emb + (size_t)idx * E_;
    u16b* dst = out + (size_t)sb * E_;
    for (int e = threadIdx.x; e < E_; e += blockDim.x)
        dst[e] = zero ? (u16b)0 : f2bf_u(src[e]);
}

__global__ void init_state_kernel(float* __restrict__ h0, u16b* __restrict__ hbf,
                                  float* __restrict__ loss) {
    int i = blockIdx.x * blockDim.x + threadIdx.x;
    if (i < B_ * H_) { h0[i] = 0.0f; hbf[i] = 0; }
    if (i == 0) loss[0] = 0.0f;
}

// GRU elementwise: r=sig(xr+hr), z=sig(xz+hz), n=tanh(xn+r*hn),
// h' = (1-z)*n + z*h.  xg carries bih (GEMM bias), gh carries bhh,
// extra = ctx_gates (decoder) or null.
__global__ void gru_update_kernel(const float* __restrict__ xg,
                                  const float* __restrict__ gh,
                                  const float* __restrict__ extra,
                                  const float* __restrict__ h_old,
                                  float* __restrict__ h_new,
                                  u16b* __restrict__ h_new_bf,
                                  float* __restrict__ outs_f32,
                                  u16b* __restrict__ outs_bf) {
    int i = blockIdx.x * blockDim.x + threadIdx.x;
    if (i >= B_ * H_) return;
    int b = i / H_, j = i % H_;
    size_t base = (size_t)b * G_;
    float xr = xg[base + j], xz = xg[base + H_ + j], xn = xg[base + 2 * H_ + j];
    if (extra) {
        xr += extra[base + j];
        xz += extra[base + H_ + j];
        xn += extra[base + 2 * H_ + j];
    }
    float hr = gh[base + j], hz = gh[base + H_ + j], hn = gh[base + 2 * H_ + j];
    float r = 1.0f / (1.0f + expf(-(xr + hr)));
    float z = 1.0f / (1.0f + expf(-(xz + hz)));
    float n = tanhf(xn + r * hn);
    float h = (1.0f - z) * n + z * h_old[i];
    h_new[i] = h;
    h_new_bf[i] = f2bf_u(h);
    if (outs_f32) { outs_f32[i] = h; outs_bf[i] = f2bf_u(h); }
}

// scores[s*B+b] = sum_h tanh(energy[s,b,h] + q[b,h]) * v[h]; one wave per (s,b)
__global__ void attn_scores_kernel(const float* __restrict__ energy,
                                   const float* __restrict__ q,
                                   const float* __restrict__ v,
                                   float* __restrict__ scores) {
    int wave = blockIdx.x * (blockDim.x >> 5) + (threadIdx.x >> 5);
    int lane = threadIdx.x & 31;
    if (wave >= S_ * B_) return;
    int b = wave % B_;
    const float* e  = energy + (size_t)wave * H_;
    const float* qb = q + (size_t)b * H_;
    float sum = 0.0f;
    for (int h = lane; h < H_; h += 32) sum += tanhf(e[h] + qb[h]) * v[h];
#pragma unroll
    for (int off = 16; off > 0; off >>= 1) sum += __shfl_xor(sum, off, 32);
    if (lane == 0) scores[wave] = sum;
}

// softmax over s (axis 0) per batch column b; grid = B, block = S
__global__ void softmax_s_kernel(const float* __restrict__ scores,
                                 float* __restrict__ w) {
    int b = blockIdx.x;
    int s = threadIdx.x;
    __shared__ float red[S_];
    float x = scores[s * B_ + b];
    red[s] = x;
    __syncthreads();
    for (int off = S_ / 2; off > 0; off >>= 1) {
        if (s < off) red[s] = fmaxf(red[s], red[s + off]);
        __syncthreads();
    }
    float m = red[0];
    __syncthreads();
    float e = expf(x - m);
    red[s] = e;
    __syncthreads();
    for (int off = S_ / 2; off > 0; off >>= 1) {
        if (s < off) red[s] += red[s + off];
        __syncthreads();
    }
    w[s * B_ + b] = e / red[0];
}

// ctx[b,h] = sum_s w[s,b] * enc_outs[s,b,h]  -> bf16
__global__ void ctx_kernel(const float* __restrict__ w,
                           const float* __restrict__ enc_outs,
                           u16b* __restrict__ ctx_bf) {
    int i = blockIdx.x * blockDim.x + threadIdx.x;
    if (i >= B_ * H_) return;
    int b = i / H_, h = i % H_;
    float acc = 0.0f;
    for (int s = 0; s < S_; ++s)
        acc += w[s * B_ + b] * enc_outs[((size_t)s * B_ + b) * H_ + h];
    ctx_bf[i] = f2bf_u(acc);
}

// Per-step masked NLL: grid = B, block = 256
__global__ void step_loss_kernel(const float* __restrict__ logits,
                                 const int* __restrict__ t,
                                 const int* __restrict__ t_len,
                                 int sStep, float* __restrict__ loss_sum) {
    int b = blockIdx.x;
    __shared__ float red[256];
    const float* row = logits + (size_t)b * V_;
    float m = -1e30f;
    for (int v = threadIdx.x; v < V_; v += blockDim.x) m = fmaxf(m, row[v]);
    red[threadIdx.x] = m;
    __syncthreads();
    for (int off = 128; off > 0; off >>= 1) {
        if (threadIdx.x < off)
            red[threadIdx.x] = fmaxf(red[threadIdx.x], red[threadIdx.x + off]);
        __syncthreads();
    }
    m = red[0];
    __syncthreads();
    float se = 0.0f;
    for (int v = threadIdx.x; v < V_; v += blockDim.x) se += expf(row[v] - m);
    red[threadIdx.x] = se;
    __syncthreads();
    for (int off = 128; off > 0; off >>= 1) {
        if (threadIdx.x < off) red[threadIdx.x] += red[threadIdx.x + off];
        __syncthreads();
    }
    if (threadIdx.x == 0) {
        if (sStep < t_len[b]) {
            int tok = t[b * S_ + sStep];
            float lp = row[tok] - m - logf(red[0]);
            atomicAdd(loss_sum, -lp);
        }
    }
}

__global__ void finalize_kernel(const float* __restrict__ loss_sum,
                                const int* __restrict__ t_len,
                                float* __restrict__ out) {
    if (blockIdx.x == 0 && threadIdx.x == 0) {
        int tot = 0;
        for (int b = 0; b < B_; ++b) {
            int l = t_len[b];
            if (l > S_) l = S_;
            tot += l;
        }
        out[0] = loss_sum[0] / (float)tot;
    }
}

// ------------------------------------------------------------------- driver

static inline void launch_gemm(const u16b* A, const u16b* W, const float* bias,
                               float* C, int M, int N, int K, hipStream_t s) {
    dim3 grid(N / 256, M / 16);
    gemm_bf16_kernel<<<grid, 128, 0, s>>>(A, W, bias, C, M, N, K);
}

extern "C" void kernel_launch(void* const* d_in, const int* in_sizes, int n_in,
                              void* d_out, int out_size, void* d_ws,
                              size_t ws_size, hipStream_t stream) {
    (void)in_sizes; (void)n_in; (void)out_size; (void)ws_size;

    const int*   s_tok    = (const int*)d_in[0];
    const int*   t_tok    = (const int*)d_in[1];
    const int*   t_len    = (const int*)d_in[2];
    const float* emb      = (const float*)d_in[3];
    const float* enc_Wih  = (const float*)d_in[4];
    const float* enc_Whh  = (const float*)d_in[5];
    const float* enc_bih  = (const float*)d_in[6];
    const float* enc_bhh  = (const float*)d_in[7];
    const float* dec_Wih  = (const float*)d_in[8];
    const float* dec_Whh  = (const float*)d_in[9];
    const float* dec_Wch  = (const float*)d_in[10];
    const float* dec_bih  = (const float*)d_in[11];
    const float* dec_bhh  = (const float*)d_in[12];
    const float* attn_W   = (const float*)d_in[13];
    const float* attn_v   = (const float*)d_in[14];
    const float* Wout     = (const float*)d_in[15];
    const float* bout     = (const float*)d_in[16];
    float*       out      = (float*)d_out;

    // ---- workspace carve-up (256B aligned)
    char*  base = (char*)d_ws;
    size_t off  = 0;
    auto carve = [&](size_t bytes) -> char* {
        char* p = base + off;
        off = (off + bytes + 255) & ~(size_t)255;
        return p;
    };
    u16b* encWih_bf = (u16b*)carve((size_t)G_ * E_ * 2);
    u16b* encWhh_bf = (u16b*)carve((size_t)G_ * H_ * 2);
    u16b* decWih_bf = (u16b*)carve((size_t)G_ * E_ * 2);
    u16b* decWhh_bf = (u16b*)carve((size_t)G_ * H_ * 2);
    u16b* decWch_bf = (u16b*)carve((size_t)G_ * H_ * 2);
    u16b* Wh_bf     = (u16b*)carve((size_t)H_ * H_ * 2);
    u16b* We_bf     = (u16b*)carve((size_t)H_ * H_ * 2);
    u16b* Wout_bf   = (u16b*)carve((size_t)V_ * H_ * 2);
    u16b* embS_bf   = (u16b*)carve((size_t)S_ * B_ * E_ * 2);
    u16b* embT_bf   = (u16b*)carve((size_t)S_ * B_ * E_ * 2);
    float* enc_xg   = (float*)carve((size_t)S_ * B_ * G_ * 4);
    float* dec_xg   = (float*)carve((size_t)S_ * B_ * G_ * 4);
    float* enc_outs = (float*)carve((size_t)S_ * B_ * H_ * 4);
    u16b* enc_outs_bf = (u16b*)carve((size_t)S_ * B_ * H_ * 2);
    float* enc_energy = (float*)carve((size_t)S_ * B_ * H_ * 4);
    float* h_pp[2];
    h_pp[0] = (float*)carve((size_t)B_ * H_ * 4);
    h_pp[1] = (float*)carve((size_t)B_ * H_ * 4);
    u16b* h_bf      = (u16b*)carve((size_t)B_ * H_ * 2);
    float* gh       = (float*)carve((size_t)B_ * G_ * 4);
    float* q        = (float*)carve((size_t)B_ * H_ * 4);
    float* scores   = (float*)carve((size_t)S_ * B_ * 4);
    float* wts      = (float*)carve((size_t)S_ * B_ * 4);
    u16b* ctx_bf    = (u16b*)carve((size_t)B_ * H_ * 2);
    float* ctx_gates = (float*)carve((size_t)B_ * G_ * 4);
    float* logits   = (float*)carve((size_t)B_ * V_ * 4);
    float* loss_sum = (float*)carve(256);

    // ---- weight / embedding conversions to bf16
    auto conv = [&](const float* src, u16b* dst, size_t n) {
        f2bf_kernel<<<(unsigned)((n + 255) / 256), 256, 0, stream>>>(src, dst, n);
    };
    conv(enc_Wih, encWih_bf, (size_t)G_ * E_);
    conv(enc_Whh, encWhh_bf, (size_t)G_ * H_);
    conv(dec_Wih, decWih_bf, (size_t)G_ * E_);
    conv(dec_Whh, decWhh_bf, (size_t)G_ * H_);
    conv(dec_Wch, decWch_bf, (size_t)G_ * H_);
    conv(Wout,    Wout_bf,   (size_t)V_ * H_);
    split_attn_kernel<<<(H_ * H_ + 255) / 256, 256, 0, stream>>>(attn_W, Wh_bf, We_bf);

    gather_embed_kernel<<<S_ * B_, 256, 0, stream>>>(emb, s_tok, embS_bf, 0);
    gather_embed_kernel<<<S_ * B_, 256, 0, stream>>>(emb, t_tok, embT_bf, 1);

    // ---- input-gate GEMMs (M=2048, N=3072, K=512) + bih bias
    launch_gemm(embS_bf, encWih_bf, enc_bih, enc_xg, S_ * B_, G_, E_, stream);
    launch_gemm(embT_bf, decWih_bf, dec_bih, dec_xg, S_ * B_, G_, E_, stream);

    // ---- init h0 = 0, loss = 0
    init_state_kernel<<<(B_ * H_ + 255) / 256, 256, 0, stream>>>(h_pp[0], h_bf, loss_sum);
    int cur = 0;

    // ---- encoder scan
    for (int s = 0; s < S_; ++s) {
        launch_gemm(h_bf, encWhh_bf, enc_bhh, gh, B_, G_, H_, stream);
        gru_update_kernel<<<(B_ * H_ + 255) / 256, 256, 0, stream>>>(
            enc_xg + (size_t)s * B_ * G_, gh, nullptr, h_pp[cur],
            h_pp[cur ^ 1], h_bf,
            enc_outs + (size_t)s * B_ * H_, enc_outs_bf + (size_t)s * B_ * H_);
        cur ^= 1;
    }

    // ---- attention energies: enc_outs @ W_e^T  (M=2048, N=1024, K=1024)
    launch_gemm(enc_outs_bf, We_bf, nullptr, enc_energy, S_ * B_, H_, H_, stream);

    // ---- decoder scan (h carries over from encoder)
    for (int s = 0; s < S_; ++s) {
        launch_gemm(h_bf, Wh_bf, nullptr, q, B_, H_, H_, stream);
        attn_scores_kernel<<<(S_ * B_ * 32) / 128, 128, 0, stream>>>(
            enc_energy, q, attn_v, scores);
        softmax_s_kernel<<<B_, S_, 0, stream>>>(scores, wts);
        ctx_kernel<<<(B_ * H_ + 255) / 256, 256, 0, stream>>>(wts, enc_outs, ctx_bf);
        launch_gemm(ctx_bf, decWch_bf, nullptr, ctx_gates, B_, G_, H_, stream);
        launch_gemm(h_bf, decWhh_bf, dec_bhh, gh, B_, G_, H_, stream);
        gru_update_kernel<<<(B_ * H_ + 255) / 256, 256, 0, stream>>>(
            dec_xg + (size_t)s * B_ * G_, gh, ctx_gates, h_pp[cur],
            h_pp[cur ^ 1], h_bf, nullptr, nullptr);
        cur ^= 1;
        launch_gemm(h_bf, Wout_bf, bout, logits, B_, V_, H_, stream);
        step_loss_kernel<<<B_, 256, 0, stream>>>(logits, t_tok, t_len, s, loss_sum);
    }

    finalize_kernel<<<1, 32, 0, stream>>>(loss_sum, t_len, out);
}
